// DotProductAttention_39676907883172
// MI455X (gfx1250) — matrix operand order repaired and checked by reading, not compiled, not run on hardware
//
#include <hip/hip_runtime.h>

// ---- problem constants (B=1, H=16, S=4096, D=64, fp32 in/out) ----
#define S_LEN   4096
#define D_DIM   64
#define H_NUM   16
#define KB      64              // keys per block iteration
#define WAVES   8
#define QROWS   128             // query rows per workgroup (16 per wave)
#define NKB     (S_LEN / KB)    // 64 key blocks

// padded LDS strides (in _Float16 elements); byte strides are multiples of 16
#define QSTR    72              // 144 B
#define KSTR    72              // 144 B
#define VSTR    72              // 144 B
#define PSTR    72              // 144 B

#define LOG2E   1.44269504088896340736f

typedef __attribute__((ext_vector_type(16))) _Float16 v16h;
typedef __attribute__((ext_vector_type(8)))  _Float16 v8h;
typedef __attribute__((ext_vector_type(4)))  _Float16 v4h;
typedef __attribute__((ext_vector_type(8)))  float    v8f;

__device__ __forceinline__ v16h cat16(v8h lo, v8h hi) {
  v16h r;
#pragma unroll
  for (int i = 0; i < 8; ++i) { r[i] = lo[i]; r[i + 8] = hi[i]; }
  return r;
}

// ---- DPP16 cross-lane helpers: butterfly within each 16-lane row ----
template <int CTRL>
__device__ __forceinline__ float dppmov(float x) {
  return __int_as_float(__builtin_amdgcn_update_dpp(
      0, __float_as_int(x), CTRL, 0xF, 0xF, true));
}
__device__ __forceinline__ float rowmax16(float v) {
  v = fmaxf(v, dppmov<0xB1>(v));    // quad_perm [1,0,3,2]  (xor 1)
  v = fmaxf(v, dppmov<0x4E>(v));    // quad_perm [2,3,0,1]  (xor 2)
  v = fmaxf(v, dppmov<0x141>(v));   // row_half_mirror      (~xor 4)
  v = fmaxf(v, dppmov<0x140>(v));   // row_mirror           (~xor 8)
  return v;
}
__device__ __forceinline__ float rowsum16(float v) {
  v += dppmov<0xB1>(v);
  v += dppmov<0x4E>(v);
  v += dppmov<0x141>(v);
  v += dppmov<0x140>(v);
  return v;
}

// B operand (32x16): lane l<16 -> col l, K {0-7,16-23}; lane l+16 -> col l, K {8-15,24-31}
__device__ __forceinline__ v16h ldsB(const char* base, int row, int strideB, int hf, int off) {
  v8h lo = *(const v8h*)(base + row * strideB + off + hf * 16);
  v8h hi = *(const v8h*)(base + row * strideB + off + hf * 16 + 32);
  return cat16(lo, hi);
}
// A operand (16x32): lane l<16 -> row l, K {0-7,16-23}; lane l+16 -> row l, K {8-15,24-31}
__device__ __forceinline__ v16h ldsA(const char* base, int rowByte, int hf, int chunkByte) {
  v8h lo = *(const v8h*)(base + rowByte + chunkByte + hf * 16);
  v8h hi = *(const v8h*)(base + rowByte + chunkByte + 32 + hf * 16);
  return cat16(lo, hi);
}

struct KVRegs { float4 k[4]; float4 v[4]; };

__device__ __forceinline__ void kvLoad(const float4* __restrict__ K4,
                                       const float4* __restrict__ V4,
                                       int tid, KVRegs& r) {
#pragma unroll
  for (int i = 0; i < 4; ++i) {
    r.k[i] = K4[tid + i * 256];
    r.v[i] = V4[tid + i * 256];
  }
}

__device__ __forceinline__ void kvStore(_Float16* __restrict__ bK,
                                        _Float16* __restrict__ bVT,
                                        int tid, const KVRegs& r) {
#pragma unroll
  for (int i = 0; i < 4; ++i) {
    int e = tid + i * 256;
    int key = e >> 4, d0 = (e * 4) & 63;
    v4h pk;
    pk[0] = (_Float16)r.k[i].x; pk[1] = (_Float16)r.k[i].y;
    pk[2] = (_Float16)r.k[i].z; pk[3] = (_Float16)r.k[i].w;
    *(v4h*)(bK + key * KSTR + d0) = pk;
    bVT[(d0 + 0) * VSTR + key] = (_Float16)r.v[i].x;
    bVT[(d0 + 1) * VSTR + key] = (_Float16)r.v[i].y;
    bVT[(d0 + 2) * VSTR + key] = (_Float16)r.v[i].z;
    bVT[(d0 + 3) * VSTR + key] = (_Float16)r.v[i].w;
  }
}

__global__ __launch_bounds__(256)
void fa_fwd_f16wmma(const float* __restrict__ Q,
                    const float* __restrict__ K,
                    const float* __restrict__ V,
                    float* __restrict__ O)
{
  __shared__ __align__(16) _Float16 sQ[QROWS * QSTR];        // [128][64]
  __shared__ __align__(16) _Float16 sKb2[2][KB * KSTR];      // ping-pong K  [64][64]
  __shared__ __align__(16) _Float16 sVb2[2][D_DIM * VSTR];   // ping-pong V^T [64][64]
  __shared__ __align__(16) _Float16 sP[WAVES * 16 * PSTR];   // per-wave 16x64 P tile

  const int tid  = threadIdx.x;
  const int wave = tid >> 5;
  const int lane = tid & 31;
  const int hf   = (lane >> 4) & 1;
  const int l15  = lane & 15;

  const int head = blockIdx.y;
  const int qblk = blockIdx.x * QROWS;
  const size_t headOff = (size_t)head * S_LEN * D_DIM;

  const float4* K4 = (const float4*)(K + headOff);   // 1024 float4 per key block
  const float4* V4 = (const float4*)(V + headOff);

  // ---- prologue: fetch KV block 0 while staging Q ----
  KVRegs kv;
  kvLoad(K4, V4, tid, kv);

  {
    const float4* Q4 = (const float4*)(Q + headOff + (size_t)qblk * D_DIM);
#pragma unroll
    for (int i = 0; i < (QROWS * D_DIM) / (256 * 4); ++i) {
      int e = tid + i * 256;
      int r = e >> 4, d0 = (e * 4) & 63;
      float4 q4 = Q4[e];
      v4h pk;
      pk[0] = (_Float16)(q4.x * LOG2E);
      pk[1] = (_Float16)(q4.y * LOG2E);
      pk[2] = (_Float16)(q4.z * LOG2E);
      pk[3] = (_Float16)(q4.w * LOG2E);
      *(v4h*)(sQ + r * QSTR + d0) = pk;
    }
  }
  kvStore(sKb2[0], sVb2[0], tid, kv);
  __syncthreads();   // Q + KV buffer 0 ready

  // ---- wave's Q rows as two 16x32 A operands, kept in registers ----
  const char* sQb = (const char*)sQ;
  const int qrowByte = (wave * 16 + l15) * (QSTR * 2);
  v16h qa0 = ldsA(sQb, qrowByte, hf, 0);
  v16h qa1 = ldsA(sQb, qrowByte, hf, 64);

  v8f o[4] = {v8f{}, v8f{}, v8f{}, v8f{}};
  float m[8], l[8];
#pragma unroll
  for (int r = 0; r < 8; ++r) { m[r] = -3.0e38f; l[r] = 0.0f; }

  _Float16* myP    = sP + wave * 16 * PSTR;
  const char* myPb = (const char*)myP;

  for (int kb = 0; kb < NKB; ++kb) {
    const int cur = kb & 1;
    const bool more = (kb + 1 < NKB);

    // issue next block's global loads early; wait lands after the WMMA work
    if (more) kvLoad(K4 + (size_t)(kb + 1) * (KB * D_DIM / 4),
                     V4 + (size_t)(kb + 1) * (KB * D_DIM / 4), tid, kv);

    const char* sKc = (const char*)sKb2[cur];
    const char* sVc = (const char*)sVb2[cur];

    // ---- S = Q x K^T : four 16x16 key tiles, K-dim = 64 (2 chunks) ----
    v8f s[4];
#pragma unroll
    for (int kt = 0; kt < 4; ++kt) {
      v8f acc = {};
      v16h b = ldsB(sKc, kt * 16 + l15, KSTR * 2, hf, 0);
      acc = __builtin_amdgcn_wmma_f32_16x16x32_f16(false, qa0, false, b, (short)0, acc, false, false);
      b = ldsB(sKc, kt * 16 + l15, KSTR * 2, hf, 64);
      acc = __builtin_amdgcn_wmma_f32_16x16x32_f16(false, qa1, false, b, (short)0, acc, false, false);
      s[kt] = acc;
    }

    // ---- online softmax over 64 columns (base-2 logits) ----
#pragma unroll
    for (int r = 0; r < 8; ++r) {
      float v = fmaxf(fmaxf(s[0][r], s[1][r]), fmaxf(s[2][r], s[3][r]));
      v = rowmax16(v);
      float mnew = fmaxf(m[r], v);

      float p0 = __builtin_amdgcn_exp2f(s[0][r] - mnew);
      float p1 = __builtin_amdgcn_exp2f(s[1][r] - mnew);
      float p2 = __builtin_amdgcn_exp2f(s[2][r] - mnew);
      float p3 = __builtin_amdgcn_exp2f(s[3][r] - mnew);

      int row = hf * 8 + r;
      myP[row * PSTR +  0 + l15] = (_Float16)p0;
      myP[row * PSTR + 16 + l15] = (_Float16)p1;
      myP[row * PSTR + 32 + l15] = (_Float16)p2;
      myP[row * PSTR + 48 + l15] = (_Float16)p3;

      float rs = (p0 + p1) + (p2 + p3);
      rs = rowsum16(rs);

      float sc = __builtin_amdgcn_exp2f(m[r] - mnew);
      l[r] = l[r] * sc + rs;
      m[r] = mnew;
      o[0][r] *= sc; o[1][r] *= sc; o[2][r] *= sc; o[3][r] *= sc;
    }

    // ---- P back through LDS into A layout (two 16x32 key chunks) ----
    v16h pa0 = ldsA(myPb, l15 * (PSTR * 2), hf, 0);
    v16h pa1 = ldsA(myPb, l15 * (PSTR * 2), hf, 64);

    // ---- O += P x V : 4 dim tiles x 2 key chunks ----
#pragma unroll
    for (int t = 0; t < 4; ++t) {
      v16h b = ldsB(sVc, t * 16 + l15, VSTR * 2, hf, 0);
      o[t] = __builtin_amdgcn_wmma_f32_16x16x32_f16(false, pa0, false, b, (short)0, o[t], false, false);
      b = ldsB(sVc, t * 16 + l15, VSTR * 2, hf, 64);
      o[t] = __builtin_amdgcn_wmma_f32_16x16x32_f16(false, pa1, false, b, (short)0, o[t], false, false);
    }

    // ---- drain in-flight globals into the other buffer; one barrier/iter ----
    if (more) kvStore(sKb2[1 - cur], sVb2[1 - cur], tid, kv);
    __syncthreads();
  }

  // ---- epilogue: normalize, coalesced stores ----
  const size_t obase = headOff + (size_t)(qblk + wave * 16) * D_DIM;
#pragma unroll
  for (int r = 0; r < 8; ++r) {
    float inv = 1.0f / l[r];
    int row = hf * 8 + r;
#pragma unroll
    for (int t = 0; t < 4; ++t)
      O[obase + (size_t)row * D_DIM + t * 16 + l15] = o[t][r] * inv;
  }
}

extern "C" void kernel_launch(void* const* d_in, const int* in_sizes, int n_in,
                              void* d_out, int out_size, void* d_ws, size_t ws_size,
                              hipStream_t stream) {
  const float* Q = (const float*)d_in[0];
  const float* K = (const float*)d_in[1];
  const float* V = (const float*)d_in[2];
  float* O = (float*)d_out;
  dim3 grid(S_LEN / QROWS, H_NUM, 1);
  fa_fwd_f16wmma<<<grid, dim3(256, 1, 1), 0, stream>>>(Q, K, V, O);
}
